// DyGraphTransformer_87342454931892
// MI455X (gfx1250) — matrix operand order repaired and checked by reading, not compiled, not run on hardware
//
#include <hip/hip_runtime.h>
#include <hip/hip_bf16.h>
#include <math.h>

// ---------------------------------------------------------------------------
// Problem constants (match reference)
// ---------------------------------------------------------------------------
static constexpr int N_NODES = 16384;
static constexpr int N_EDGES = 131072;
static constexpr int F_IN    = 256;
static constexpr int HID     = 512;
static constexpr int HALF    = 256;
static constexpr int HEADS   = 8;
static constexpr int DHEAD   = 32;
static constexpr int LAYERS  = 4;

// ---------------------------------------------------------------------------
// CDNA5 WMMA types
// ---------------------------------------------------------------------------
typedef __attribute__((ext_vector_type(16))) __bf16 v16bf;
typedef __attribute__((ext_vector_type(8)))  float  v8f;

union ABf { uint4 u[2]; v16bf v; };

__device__ __forceinline__ unsigned short f2bf(float f) {
    unsigned u = __float_as_uint(f);
    unsigned r = u + 0x7FFFu + ((u >> 16) & 1u);   // round-to-nearest-even
    return (unsigned short)(r >> 16);
}
__device__ __forceinline__ unsigned pk2bf(float a, float b) {
    return (unsigned)f2bf(a) | ((unsigned)f2bf(b) << 16);
}

__device__ __forceinline__ float waveSum(float v) {
    #pragma unroll
    for (int o = 16; o > 0; o >>= 1) v += __shfl_xor(v, o, 32);
    return v;
}

// order-preserving float<->uint mapping for atomic segment-max
__device__ __forceinline__ unsigned fkey(float f) {
    unsigned u = __float_as_uint(f);
    return (u & 0x80000000u) ? ~u : (u | 0x80000000u);
}
__device__ __forceinline__ float kinv(unsigned k) {
    unsigned u = (k & 0x80000000u) ? (k & 0x7FFFFFFFu) : ~k;
    return __uint_as_float(u);
}

// ---------------------------------------------------------------------------
// bf16 WMMA GEMM: C[M,Nc] = op(A[M,K] @ B[K,Nc] + bias [, +res])
// MODE 0: +bias   MODE 1: +bias then +res (residual)   MODE 2: gelu(+bias)
// Block: 128 threads = 4 waves; block tile 128x64; each wave 32x64; K step 32.
// Per K-step per wave: 2 A-frags + 4 B-frags (12 x ds_load_b128) -> 8 WMMA.
// ---------------------------------------------------------------------------
template<int MODE>
__global__ __launch_bounds__(128)
void gemm_bf16_wmma(const float* __restrict__ A, const float* __restrict__ B,
                    const float* __restrict__ bias, const float* __restrict__ res,
                    float* __restrict__ C, int M, int K, int Nc)
{
    __shared__ unsigned short lA[128][40];  // [m][k], 80B rows (16B aligned frags)
    __shared__ unsigned short lB[64][40];   // [n][k] (B stored transposed)

    const int tid  = threadIdx.x;
    const int lane = tid & 31;
    const int wave = tid >> 5;
    const int m0   = blockIdx.y * 128;
    const int n0   = blockIdx.x * 64;

    v8f acc[2][4];
    #pragma unroll
    for (int s = 0; s < 2; ++s)
        #pragma unroll
        for (int t = 0; t < 4; ++t) acc[s][t] = (v8f){};

    const int ar  = tid >> 3;            // 0..15  (A stage row group)
    const int ac4 = (tid & 7) << 2;      // 0..28  (A stage col)
    const int bkr = tid >> 4;            // 0..7   (B stage k row)
    const int bc4 = (tid & 15) << 2;     // 0..60  (B stage col)

    for (int kk = 0; kk < K; kk += 32) {
        // ---- prefetch next K-step tiles into cache (global_prefetch_b8) ----
        if (kk + 32 < K) {
            __builtin_prefetch(A + (size_t)(m0 + ar) * K + kk + 32 + ac4, 0, 1);
            __builtin_prefetch(B + (size_t)(kk + 32 + bkr) * Nc + n0 + bc4, 0, 1);
        }
        // ---- stage A tile 128x32 (f32 -> bf16, packed b64 stores) ----
        {
            const float* ap = A + (size_t)(m0 + ar) * K + kk + ac4;
            #pragma unroll
            for (int p = 0; p < 8; ++p) {
                float4 f = *(const float4*)(ap + (size_t)p * 16 * K);
                uint2 pk; pk.x = pk2bf(f.x, f.y); pk.y = pk2bf(f.z, f.w);
                *(uint2*)&lA[p * 16 + ar][ac4] = pk;
            }
        }
        // ---- stage B tile 32x64 transposed -> lB[n][k] ----
        {
            const float* bp = B + (size_t)(kk + bkr) * Nc + n0 + bc4;
            #pragma unroll
            for (int p = 0; p < 4; ++p) {
                float4 f = *(const float4*)(bp + (size_t)p * 8 * Nc);
                const int krow = p * 8 + bkr;
                lB[bc4 + 0][krow] = f2bf(f.x);
                lB[bc4 + 1][krow] = f2bf(f.y);
                lB[bc4 + 2][krow] = f2bf(f.z);
                lB[bc4 + 3][krow] = f2bf(f.w);
            }
        }
        __syncthreads();

        // ---- A fragments (ISA 16-bit A 16x32 layout), 2 row-tiles per wave ----
        const int mr = lane & 15;
        const int ka = (lane < 16) ? 0 : 8;
        const int kb = (lane < 16) ? 0 : 8;
        ABf af[2];
        #pragma unroll
        for (int s = 0; s < 2; ++s) {
            af[s].u[0] = *(const uint4*)&lA[wave * 32 + s * 16 + mr][ka];
            af[s].u[1] = *(const uint4*)&lA[wave * 32 + s * 16 + mr][ka + 16];
        }
        // ---- B fragments + WMMA (each B frag reused for both row-tiles) ----
        #pragma unroll
        for (int t = 0; t < 4; ++t) {
            ABf bf_;
            bf_.u[0] = *(const uint4*)&lB[t * 16 + mr][kb];
            bf_.u[1] = *(const uint4*)&lB[t * 16 + mr][kb + 16];
            acc[0][t] = __builtin_amdgcn_wmma_f32_16x16x32_bf16(
                            false, af[0].v, false, bf_.v, (short)0, acc[0][t], false, false);
            acc[1][t] = __builtin_amdgcn_wmma_f32_16x16x32_bf16(
                            false, af[1].v, false, bf_.v, (short)0, acc[1][t], false, false);
        }
        __syncthreads();
    }

    // ---- epilogue: C/D layout (lane = N col, VGPR r = M row) ----
    const int n_in = lane & 15;
    const int mofs = (lane < 16) ? 0 : 8;
    #pragma unroll
    for (int s = 0; s < 2; ++s) {
        #pragma unroll
        for (int t = 0; t < 4; ++t) {
            const int col = n0 + t * 16 + n_in;
            const float bb = bias ? bias[col] : 0.f;
            #pragma unroll
            for (int r = 0; r < 8; ++r) {
                const int grow = m0 + wave * 32 + s * 16 + r + mofs;
                float val = acc[s][t][r] + bb;
                if (MODE == 2) val = 0.5f * val * (1.f + erff(val * 0.70710678118654752440f));
                if (MODE == 1) val += res[(size_t)grow * Nc + col];
                C[(size_t)grow * Nc + col] = val;
            }
        }
    }
}

// ---------------------------------------------------------------------------
// LayerNorm over rows of 256 (wave per row)
// ---------------------------------------------------------------------------
__global__ __launch_bounds__(256)
void layernorm_rows(const float* __restrict__ X, const float* __restrict__ g,
                    const float* __restrict__ b, float* __restrict__ Y, int rows)
{
    const int wid = blockIdx.x * 8 + (threadIdx.x >> 5);
    if (wid >= rows) return;
    const int lane = threadIdx.x & 31;
    const float* x = X + (size_t)wid * HALF + lane * 8;
    float v[8]; float s = 0.f, sq = 0.f;
    #pragma unroll
    for (int j = 0; j < 8; ++j) { v[j] = x[j]; s += v[j]; sq += v[j] * v[j]; }
    #pragma unroll
    for (int o = 16; o > 0; o >>= 1) { s += __shfl_xor(s, o, 32); sq += __shfl_xor(sq, o, 32); }
    const float mean = s * (1.f / HALF);
    const float var  = sq * (1.f / HALF) - mean * mean;
    const float inv  = rsqrtf(var + 1e-5f);
    float* y = Y + (size_t)wid * HALF + lane * 8;
    #pragma unroll
    for (int j = 0; j < 8; ++j) {
        const int d = lane * 8 + j;
        y[j] = (v[j] - mean) * inv * g[d] + b[d];
    }
}

// ---------------------------------------------------------------------------
// node token embeddings (max_norm lookups, mean over W=2), add to h, split
// ---------------------------------------------------------------------------
__global__ __launch_bounds__(256)
void add_node_te_split(const float* __restrict__ h, const int* __restrict__ enc,
                       const float* __restrict__ emb,
                       float* __restrict__ x1, float* __restrict__ x2)
{
    const int node = blockIdx.x * 8 + (threadIdx.x >> 5);
    if (node >= N_NODES) return;
    const int lane = threadIdx.x & 31;
    const int i0 = enc[node * 2 + 0], i1 = enc[node * 2 + 1];
    const float* r0 = emb + (size_t)i0 * HID + lane * 16;
    const float* r1 = emb + (size_t)i1 * HID + lane * 16;
    float a0[16], a1[16]; float ss0 = 0.f, ss1 = 0.f;
    #pragma unroll
    for (int j = 0; j < 16; ++j) {
        a0[j] = r0[j]; ss0 += a0[j] * a0[j];
        a1[j] = r1[j]; ss1 += a1[j] * a1[j];
    }
    ss0 = waveSum(ss0); ss1 = waveSum(ss1);
    const float s0 = fminf(1.f, 1.f / fmaxf(sqrtf(ss0), 1e-7f));
    const float s1 = fminf(1.f, 1.f / fmaxf(sqrtf(ss1), 1e-7f));
    #pragma unroll
    for (int j = 0; j < 16; ++j) {
        const int d = lane * 16 + j;
        const float val = h[(size_t)node * HID + d] + 0.5f * (a0[j] * s0 + a1[j] * s1);
        if (d < HALF) x1[(size_t)node * HALF + d] = val;
        else          x2[(size_t)node * HALF + (d - HALF)] = val;
    }
}

// ---------------------------------------------------------------------------
// attention bias per edge: mean(maxnorm(edge_emb)) @ Wedge + maxnorm(dist) @ Wdist
// ---------------------------------------------------------------------------
__global__ __launch_bounds__(256)
void edge_bias_kernel(const int* __restrict__ ee, const int* __restrict__ ed,
                      const float* __restrict__ eemb, const float* __restrict__ demb,
                      const float* __restrict__ We, const float* __restrict__ be,
                      const float* __restrict__ Wd, const float* __restrict__ bd,
                      float* __restrict__ biasEH)
{
    const int e = blockIdx.x * 8 + (threadIdx.x >> 5);
    if (e >= N_EDGES) return;
    const int lane = threadIdx.x & 31;
    const int i0 = ee[e * 2], i1 = ee[e * 2 + 1], di = ed[e];
    const float* r0 = eemb + (size_t)i0 * HID + lane * 16;
    const float* r1 = eemb + (size_t)i1 * HID + lane * 16;
    const float* rd = demb + (size_t)di * HID + lane * 16;
    float a0[16], a1[16], b[16]; float ss0 = 0.f, ss1 = 0.f, ssd = 0.f;
    #pragma unroll
    for (int j = 0; j < 16; ++j) {
        a0[j] = r0[j]; ss0 += a0[j] * a0[j];
        a1[j] = r1[j]; ss1 += a1[j] * a1[j];
        b[j]  = rd[j]; ssd += b[j]  * b[j];
    }
    ss0 = waveSum(ss0); ss1 = waveSum(ss1); ssd = waveSum(ssd);
    const float s0 = fminf(1.f, 1.f / fmaxf(sqrtf(ss0), 1e-7f));
    const float s1 = fminf(1.f, 1.f / fmaxf(sqrtf(ss1), 1e-7f));
    const float sd = fminf(1.f, 1.f / fmaxf(sqrtf(ssd), 1e-7f));
    #pragma unroll
    for (int j = 0; j < 16; ++j) {
        a0[j] = 0.5f * (a0[j] * s0 + a1[j] * s1);   // mean of maxnorm rows
        b[j]  = b[j] * sd;
    }
    #pragma unroll
    for (int hh = 0; hh < HEADS; ++hh) {
        float p = 0.f;
        #pragma unroll
        for (int j = 0; j < 16; ++j) {
            const int d = lane * 16 + j;
            p += a0[j] * We[(size_t)d * HEADS + hh] + b[j] * Wd[(size_t)d * HEADS + hh];
        }
        p = waveSum(p);
        if (lane == 0) biasEH[(size_t)e * HEADS + hh] = p + be[hh] + bd[hh];
    }
}

// ---------------------------------------------------------------------------
// edge attention passes
// ---------------------------------------------------------------------------
__global__ __launch_bounds__(256)
void attn_clear(float* __restrict__ accum, unsigned* __restrict__ mkey,
                float* __restrict__ denom)
{
    const size_t i = (size_t)blockIdx.x * 256 + threadIdx.x;
    accum[i] = 0.f;
    if (i < (size_t)N_NODES * HEADS) {
        mkey[i]  = fkey(-INFINITY);   // 0x007FFFFF
        denom[i] = 0.f;
    }
}

__global__ __launch_bounds__(256)
void attn_scores(const float* __restrict__ q, const float* __restrict__ k,
                 const float* __restrict__ biasEH,
                 const int* __restrict__ row, const int* __restrict__ col,
                 float* __restrict__ scores, unsigned* __restrict__ mkey)
{
    const int e = blockIdx.x * 8 + (threadIdx.x >> 5);
    if (e >= N_EDGES) return;
    const int lane = threadIdx.x & 31;
    const int h = lane >> 2, sub = lane & 3;
    const int r = row[e], c = col[e];
    const float* qp = q + (size_t)r * HALF + h * DHEAD + sub * 8;
    const float* kp = k + (size_t)c * HALF + h * DHEAD + sub * 8;
    float p = 0.f;
    #pragma unroll
    for (int j = 0; j < 8; ++j) p += qp[j] * kp[j];
    p += __shfl_xor(p, 1, 32);
    p += __shfl_xor(p, 2, 32);
    if (sub == 0) {
        const float s = p * 0.17677669529663687f /* 32^-0.5 */ + biasEH[(size_t)e * HEADS + h];
        scores[(size_t)e * HEADS + h] = s;
        atomicMax(&mkey[(size_t)r * HEADS + h], fkey(s));
    }
}

__global__ __launch_bounds__(256)
void attn_accum(const float* __restrict__ v, const float* __restrict__ scores,
                const unsigned* __restrict__ mkey,
                const int* __restrict__ row, const int* __restrict__ col,
                float* __restrict__ accum, float* __restrict__ denom)
{
    const int e = blockIdx.x * 8 + (threadIdx.x >> 5);
    if (e >= N_EDGES) return;
    const int lane = threadIdx.x & 31;
    const int h = lane >> 2, sub = lane & 3;
    const int r = row[e], c = col[e];
    const float m = kinv(mkey[(size_t)r * HEADS + h]);
    const float a = __expf(scores[(size_t)e * HEADS + h] - m);
    const float* vp = v + (size_t)c * HALF + h * DHEAD + sub * 8;
    float* op = accum + (size_t)r * HALF + h * DHEAD + sub * 8;
    #pragma unroll
    for (int j = 0; j < 8; ++j) atomicAdd(&op[j], a * vp[j]);
    if (sub == 0) atomicAdd(&denom[(size_t)r * HEADS + h], a);
}

__global__ __launch_bounds__(256)
void attn_norm(float* __restrict__ accum, const float* __restrict__ denom)
{
    const size_t i = (size_t)blockIdx.x * 256 + threadIdx.x;
    const int n = (int)(i >> 8);
    const int h = (int)((i >> 5) & 7);
    accum[i] /= fmaxf(denom[(size_t)n * HEADS + h], 1e-9f);
}

__global__ __launch_bounds__(256)
void concat_out(const float* __restrict__ x1, const float* __restrict__ x2,
                float* __restrict__ out)
{
    const size_t i = (size_t)blockIdx.x * 256 + threadIdx.x;
    const int n = (int)(i >> 9);
    const int d = (int)(i & 511);
    out[i] = (d < HALF) ? x1[(size_t)n * HALF + d]
                        : x2[(size_t)n * HALF + (d - HALF)];
}

// ---------------------------------------------------------------------------
// launch
// ---------------------------------------------------------------------------
extern "C" void kernel_launch(void* const* d_in, const int* in_sizes, int n_in,
                              void* d_out, int out_size, void* d_ws, size_t ws_size,
                              hipStream_t stream)
{
    const float* x          = (const float*)d_in[0];
    const int*   node_enc   = (const int*)  d_in[1];
    const int*   edge_enc   = (const int*)  d_in[2];
    const int*   edge_dist  = (const int*)  d_in[3];
    const int*   srow       = (const int*)  d_in[4];
    const int*   scol       = (const int*)  d_in[5];
    const float* node_emb   = (const float*)d_in[6];
    const float* edge_emb   = (const float*)d_in[7];
    const float* dist_emb   = (const float*)d_in[8];
    const float* Wfeat      = (const float*)d_in[9];
    const float* bfeat      = (const float*)d_in[10];
    const float* Wedge      = (const float*)d_in[11];
    const float* bedge      = (const float*)d_in[12];
    const float* Wdist      = (const float*)d_in[13];
    const float* bdist      = (const float*)d_in[14];
    const float* ln1_g      = (const float*)d_in[15];
    const float* ln1_b      = (const float*)d_in[16];
    const float* Wq         = (const float*)d_in[17];
    const float* bq         = (const float*)d_in[18];
    const float* Wk         = (const float*)d_in[19];
    const float* bk         = (const float*)d_in[20];
    const float* Wv         = (const float*)d_in[21];
    const float* bv         = (const float*)d_in[22];
    const float* Wo         = (const float*)d_in[23];
    const float* bo         = (const float*)d_in[24];
    const float* ln2_g      = (const float*)d_in[25];
    const float* ln2_b      = (const float*)d_in[26];
    const float* W1         = (const float*)d_in[27];
    const float* b1         = (const float*)d_in[28];
    const float* W2         = (const float*)d_in[29];
    const float* b2         = (const float*)d_in[30];

    // workspace layout (floats)
    float* ws = (float*)d_ws;
    const size_t NH = (size_t)N_NODES * HALF;   // 16384*256
    float*    x1     = ws + 0 * NH;
    float*    x2     = ws + 1 * NH;
    float*    lnbuf  = ws + 2 * NH;
    float*    qbuf   = ws + 3 * NH;             // h[N,512] aliases q|k
    float*    kbuf   = ws + 4 * NH;
    float*    hbuf   = qbuf;                    // [N,512]
    float*    vbuf   = ws + 5 * NH;
    float*    accum  = ws + 6 * NH;             // also FFN t1
    float*    biasEH = ws + 7 * NH;                         // E*8
    float*    scores = biasEH + (size_t)N_EDGES * HEADS;    // E*8
    unsigned* mkey   = (unsigned*)(scores + (size_t)N_EDGES * HEADS); // N*8
    float*    denom  = (float*)(mkey + (size_t)N_NODES * HEADS);      // N*8

    const dim3 blk128(128), blk256(256);
    const int  ROWB  = N_NODES / 8;     // wave-per-row blocks
    const int  EDGB  = N_EDGES / 8;     // wave-per-edge blocks
    const dim3 gemm512(HID / 64, N_NODES / 128);
    const dim3 gemm256(HALF / 64, N_NODES / 128);

    // h = x @ Wfeat + bfeat      [N,512]
    gemm_bf16_wmma<0><<<gemm512, blk128, 0, stream>>>(x, Wfeat, bfeat, nullptr,
                                                      hbuf, N_NODES, F_IN, HID);
    // attn bias per edge (independent of h)
    edge_bias_kernel<<<EDGB, blk256, 0, stream>>>(edge_enc, edge_dist, edge_emb, dist_emb,
                                                  Wedge, bedge, Wdist, bdist, biasEH);
    // h += node token embeddings; split into x1 | x2  (frees hbuf for q/k reuse)
    add_node_te_split<<<ROWB, blk256, 0, stream>>>(hbuf, node_enc, node_emb, x1, x2);

    for (int l = 0; l < LAYERS; ++l) {
        const size_t wOff = (size_t)l * HALF * HALF;
        const size_t vOff = (size_t)l * HALF;

        // f_in = LN(x2)
        layernorm_rows<<<ROWB, blk256, 0, stream>>>(x2, ln1_g + vOff, ln1_b + vOff,
                                                    lnbuf, N_NODES);
        // q, k, v projections
        gemm_bf16_wmma<0><<<gemm256, blk128, 0, stream>>>(lnbuf, Wq + wOff, bq + vOff,
                                                          nullptr, qbuf, N_NODES, HALF, HALF);
        gemm_bf16_wmma<0><<<gemm256, blk128, 0, stream>>>(lnbuf, Wk + wOff, bk + vOff,
                                                          nullptr, kbuf, N_NODES, HALF, HALF);
        gemm_bf16_wmma<0><<<gemm256, blk128, 0, stream>>>(lnbuf, Wv + wOff, bv + vOff,
                                                          nullptr, vbuf, N_NODES, HALF, HALF);
        // edge-list attention (segment max / sum via float atomics)
        attn_clear <<<N_NODES, blk256, 0, stream>>>(accum, mkey, denom);
        attn_scores<<<EDGB,    blk256, 0, stream>>>(qbuf, kbuf, biasEH, srow, scol,
                                                    scores, mkey);
        attn_accum <<<EDGB,    blk256, 0, stream>>>(vbuf, scores, mkey, srow, scol,
                                                    accum, denom);
        attn_norm  <<<N_NODES, blk256, 0, stream>>>(accum, denom);
        // y1 = x1 + accum @ Wo + bo    (residual GEMM, in-place on x1)
        gemm_bf16_wmma<1><<<gemm256, blk128, 0, stream>>>(accum, Wo + wOff, bo + vOff,
                                                          x1, x1, N_NODES, HALF, HALF);
        // g_in = LN(y1)
        layernorm_rows<<<ROWB, blk256, 0, stream>>>(x1, ln2_g + vOff, ln2_b + vOff,
                                                    lnbuf, N_NODES);
        // t1 = gelu(g_in @ W1 + b1)   (reuse accum as t1)
        gemm_bf16_wmma<2><<<gemm256, blk128, 0, stream>>>(lnbuf, W1 + wOff, b1 + vOff,
                                                          nullptr, accum, N_NODES, HALF, HALF);
        // y2 = x2 + t1 @ W2 + b2      (residual GEMM, in-place on x2)
        gemm_bf16_wmma<1><<<gemm256, blk128, 0, stream>>>(accum, W2 + wOff, b2 + vOff,
                                                          x2, x2, N_NODES, HALF, HALF);
    }

    // out = concat(x1, x2)  [N,512]
    concat_out<<<(N_NODES * HID) / 256, blk256, 0, stream>>>(x1, x2, (float*)d_out);
}
